// EedgePath_MPNN_58256936403397
// MI455X (gfx1250) — matrix operand order repaired and checked by reading, not compiled, not run on hardware
//
#include <hip/hip_runtime.h>
#include <math.h>

#define N_NODES 50000
#define N_EDGES 800000
#define DD      64
#define LAYERS  3

typedef __attribute__((ext_vector_type(16))) __bf16 v16bf;
typedef __attribute__((ext_vector_type(8)))  float  v8f;

// ---- fragment index maps (CDNA5 ISA 7.12.2, wave32) ----
// A (16-bit, 16x32, MxK): lane m=lane&15, khalf=lane>>4.
//   j=0..7  -> K = j      + khalf*8
//   j=8..15 -> K = (j+8)  + khalf*8   (i.e. 16 + (j-8) + khalf*8)
__device__ __forceinline__ int kmapA(int j, int khalf) {
  return (j < 8 ? j : j + 8) + khalf * 8;
}
// B (16-bit, 32x16, KxN): lane n=lane&15; same K map as A (j<8 -> j+khalf*8, j>=8 -> 16+(j-8)+khalf*8).
__device__ __forceinline__ int kmapB(int j, int khalf) {
  return (j < 8 ? j : j + 8) + khalf * 8;
}

__device__ __forceinline__ v8f wmma_bf16(v16bf a, v16bf b, v8f c) {
  return __builtin_amdgcn_wmma_f32_16x16x32_bf16(
      /*neg_a=*/false, a, /*neg_b=*/false, b,
      /*c_mod=*/(short)0, c, /*reuse_a=*/false, /*reuse_b=*/false);
}

// |x| via opaque inline asm. Source-level fabs/fmax(d,-d) gets canonicalized
// to llvm.fabs on vectorized (v2f32) LDS loads, which this gfx1250 backend
// cannot select. Inline asm blocks that re-combination and emits exactly one
// v_and_b32 (VOP2, literal in src0).
__device__ __forceinline__ float abs_f32(float x) {
  float r;
  asm("v_and_b32 %0, 0x7fffffff, %1" : "=v"(r) : "v"(x));
  return r;
}

// ------------------------------------------------------------------
__global__ void zero_f32(float* __restrict__ p, int n) {
  int t = blockIdx.x * blockDim.x + threadIdx.x;
  if (t < n) p[t] = 0.0f;
}

__global__ void compute_deg(const int* __restrict__ trg, float* __restrict__ deg, int E) {
  int t = blockIdx.x * blockDim.x + threadIdx.x;
  if (t < E) atomicAdd(&deg[trg[t]], 1.0f);
}

// Pre-swizzle all weights into bf16 B-fragment layout (one-time, tiny).
// Fragment flat index: (((layer*KS + ks)*4 + nt)*32 + lane)*16 + j
__global__ void prep_weights(const float* __restrict__ Wself,
                             const float* __restrict__ Wnbr,
                             const float* __restrict__ Wfc,
                             __bf16* __restrict__ fself,
                             __bf16* __restrict__ fnbr,
                             __bf16* __restrict__ ffc,
                             float* __restrict__ w256) {
  const int NODE = LAYERS * 2 * 4 * 32 * 16;  // 12288 per matrix
  const int FC   = LAYERS * 8 * 4 * 32 * 16;  // 49152
  int t = blockIdx.x * blockDim.x + threadIdx.x;
  if (t < 2 * NODE) {
    int q = t % NODE;
    bool isn = (t >= NODE);
    int j = q & 15, lane = (q >> 4) & 31, nt = (q >> 9) & 3, ks = (q >> 11) & 1, ly = q >> 12;
    int k = ks * 32 + kmapB(j, lane >> 4);
    int n = nt * 16 + (lane & 15);
    const float* W = isn ? Wnbr : Wself;
    float v = W[((size_t)ly * DD + k) * DD + n];
    (isn ? fnbr : fself)[q] = (__bf16)v;
  } else if (t < 2 * NODE + FC) {
    int q = t - 2 * NODE;
    int j = q & 15, lane = (q >> 4) & 31, nt = (q >> 9) & 3, ks = (q >> 11) & 7, ly = q >> 14;
    int k = ks * 32 + kmapB(j, lane >> 4);
    int n = nt * 16 + (lane & 15);
    ffc[q] = (__bf16)Wfc[((size_t)ly * 257 + k) * DD + n];
  } else if (t < 2 * NODE + FC + LAYERS * DD) {
    int q = t - 2 * NODE - FC;
    int ly = q >> 6, n = q & 63;
    w256[q] = Wfc[((size_t)ly * 257 + 256) * DD + n];
  }
}

// Scatter-add x[src] rows into agg[trg]. agg fits in L2 -> atomics resolve there.
// One wave handles 2 edges: 16 lanes x float4 = one 256B row each.
__global__ void scatter_agg(const float* __restrict__ x,
                            const int* __restrict__ src,
                            const int* __restrict__ trg,
                            float* __restrict__ agg, int E) {
  int wid  = blockIdx.x * (blockDim.x >> 5) + (threadIdx.x >> 5);
  int lane = threadIdx.x & 31;
  int e = wid * 2 + (lane >> 4);
  if (e >= E) return;
  int s = src[e], t = trg[e];
  int f0 = (lane & 15) * 4;
  const float4 v = *(const float4*)(x + (size_t)s * DD + f0);
  float* a = agg + (size_t)t * DD + f0;
  atomicAdd(a + 0, v.x);
  atomicAdd(a + 1, v.y);
  atomicAdd(a + 2, v.z);
  atomicAdd(a + 3, v.w);
}

// x_pre = x@W_self + (agg/deg)@W_nbr + b ; x_relu = relu(x_pre)
// One wave per 16-row tile: 2 K-steps x 4 N-tiles x 2 matrices = 16 WMMAs.
__global__ __launch_bounds__(128) void node_gemm(
    const float* __restrict__ x, const float* __restrict__ agg,
    const float* __restrict__ deg,
    const __bf16* __restrict__ wself, const __bf16* __restrict__ wnbr,
    const float* __restrict__ bconv,
    float* __restrict__ xpre, float* __restrict__ xrelu, int ntiles) {
  int wv = blockIdx.x * (blockDim.x >> 5) + (threadIdx.x >> 5);
  if (wv >= ntiles) return;
  int lane = threadIdx.x & 31;
  int m = lane & 15, khalf = lane >> 4;
  int row = wv * 16 + m;
  float dinv = 1.0f / fmaxf(deg[row], 1.0f);
  const float* xr = x + (size_t)row * DD;
  const float* ar = agg + (size_t)row * DD;

  v8f acc[4] = {};
#pragma unroll
  for (int ks = 0; ks < 2; ++ks) {
    v16bf ax, ag;
#pragma unroll
    for (int j = 0; j < 16; ++j) {
      int k = ks * 32 + kmapA(j, khalf);
      ax[j] = (__bf16)xr[k];
      ag[j] = (__bf16)(ar[k] * dinv);
    }
#pragma unroll
    for (int nt = 0; nt < 4; ++nt) {
      v16bf bs = *(const v16bf*)(wself + (size_t)(((ks * 4 + nt) * 32 + lane) * 16));
      v16bf bn = *(const v16bf*)(wnbr  + (size_t)(((ks * 4 + nt) * 32 + lane) * 16));
      acc[nt] = wmma_bf16(ax, bs, acc[nt]);
      acc[nt] = wmma_bf16(ag, bn, acc[nt]);
    }
  }
  // C layout: VGPR r holds rows (r, r+8); lanes 0-15 -> M=r, lanes 16-31 -> M=r+8; N=lane&15.
#pragma unroll
  for (int nt = 0; nt < 4; ++nt) {
    int nc = nt * 16 + m;
    float bv = bconv[nc];
#pragma unroll
    for (int r = 0; r < 8; ++r) {
      int mm = r + khalf * 8;
      float v = acc[nt][r] + bv;
      size_t o = (size_t)(wv * 16 + mm) * DD + nc;
      xpre[o]  = v;
      xrelu[o] = fmaxf(v, 0.0f);
    }
  }
}

// Edge MLP: ef_out = [ef | x_src | x_trg | |x_src-x_trg| | sim] @ W_fc + b_fc
// One wave per 16-edge tile; 8 bf16 K-steps (K=256) + rank-1 fp32 sim term.
__global__ __launch_bounds__(128) void edge_update(
    const float* __restrict__ xpre,
    const int* __restrict__ src, const int* __restrict__ trg,
    const float* __restrict__ ef_in,
    const __bf16* __restrict__ wfc, const float* __restrict__ w256,
    const float* __restrict__ bfc,
    float* __restrict__ ef_out, int E) {
  __shared__ float s_src[4][16][65];  // stride-65 padding -> conflict-free column reads
  __shared__ float s_trg[4][16][65];
  __shared__ float s_ef [4][16][65];
  __shared__ float s_sim[4][16];

  int w    = threadIdx.x >> 5;
  int lane = threadIdx.x & 31;
  int tile = blockIdx.x * 4 + w;     // grid sized exactly: E/16/4 blocks
  int ebase = tile * 16;
  int m = lane & 15, khalf = lane >> 4;

  // Stage 16 edges: 2 rows per iter (half-wave each), 16 lanes x float4 per row.
  int f0 = (lane & 15) * 4;
#pragma unroll
  for (int r = 0; r < 16; r += 2) {
    int mm = r + khalf;
    int e = ebase + mm;
    int s = src[e], t = trg[e];
    float4 vs = *(const float4*)(xpre + (size_t)s * DD + f0);
    float4 vt = *(const float4*)(xpre + (size_t)t * DD + f0);
    float4 ve = *(const float4*)(ef_in + (size_t)e * DD + f0);
    s_src[w][mm][f0 + 0] = vs.x; s_src[w][mm][f0 + 1] = vs.y;
    s_src[w][mm][f0 + 2] = vs.z; s_src[w][mm][f0 + 3] = vs.w;
    s_trg[w][mm][f0 + 0] = vt.x; s_trg[w][mm][f0 + 1] = vt.y;
    s_trg[w][mm][f0 + 2] = vt.z; s_trg[w][mm][f0 + 3] = vt.w;
    s_ef [w][mm][f0 + 0] = ve.x; s_ef [w][mm][f0 + 1] = ve.y;
    s_ef [w][mm][f0 + 2] = ve.z; s_ef [w][mm][f0 + 3] = ve.w;
  }
  __syncthreads();

  // cosine sim: each half-wave covers 32 features, combine across halves.
  float dot = 0.0f, q1 = 0.0f, q2 = 0.0f;
#pragma unroll
  for (int ff = 0; ff < 32; ++ff) {
    int f = khalf * 32 + ff;
    float a = s_src[w][m][f], b = s_trg[w][m][f];
    dot += a * b; q1 += a * a; q2 += b * b;
  }
  dot += __shfl_xor(dot, 16, 32);
  q1  += __shfl_xor(q1, 16, 32);
  q2  += __shfl_xor(q2, 16, 32);
  float sim = dot / fmaxf(sqrtf(q1) * sqrtf(q2), 1e-8f);
  if (lane < 16) s_sim[w][m] = sim;
  __syncthreads();

  v8f acc[4] = {};
#pragma unroll
  for (int ks = 0; ks < 8; ++ks) {
    const int region = ks >> 1;           // 0:ef 1:x_src 2:x_trg 3:|diff| (compile-time)
    const int fbase  = (ks & 1) * 32;
    v16bf a;
#pragma unroll
    for (int j = 0; j < 16; ++j) {
      int f = fbase + kmapA(j, khalf);
      float v;
      if (region == 0)      v = s_ef [w][m][f];
      else if (region == 1) v = s_src[w][m][f];
      else if (region == 2) v = s_trg[w][m][f];
      else                  v = abs_f32(s_src[w][m][f] - s_trg[w][m][f]);
      a[j] = (__bf16)v;
    }
#pragma unroll
    for (int nt = 0; nt < 4; ++nt) {
      v16bf b = *(const v16bf*)(wfc + (size_t)(((ks * 4 + nt) * 32 + lane) * 16));
      acc[nt] = wmma_bf16(a, b, acc[nt]);
    }
  }

  // epilogue: + bias + sim * W_fc[256,:]  (rank-1, fp32 exact)
#pragma unroll
  for (int nt = 0; nt < 4; ++nt) {
    int nc = nt * 16 + m;
    float bv = bfc[nc];
    float wv = w256[nc];
#pragma unroll
    for (int r = 0; r < 8; ++r) {
      int mm = r + khalf * 8;
      float v = acc[nt][r] + bv + s_sim[w][mm] * wv;
      ef_out[(size_t)(ebase + mm) * DD + nc] = v;
    }
  }
  (void)E;
}

// ------------------------------------------------------------------
extern "C" void kernel_launch(void* const* d_in, const int* in_sizes, int n_in,
                              void* d_out, int out_size, void* d_ws, size_t ws_size,
                              hipStream_t stream) {
  (void)in_sizes; (void)n_in; (void)out_size; (void)ws_size;
  const float* x_in   = (const float*)d_in[0];
  const int*   eidx   = (const int*)d_in[1];
  const float* ef0    = (const float*)d_in[2];
  const float* Wself  = (const float*)d_in[3];
  const float* Wnbr   = (const float*)d_in[4];
  const float* bconv  = (const float*)d_in[5];
  const float* Wfc    = (const float*)d_in[6];
  const float* bfc    = (const float*)d_in[7];
  const int* src = eidx;
  const int* trg = eidx + N_EDGES;
  float* out = (float*)d_out;

  char* ws = (char*)d_ws;
  size_t off = 0;
  auto take = [&](size_t bytes) -> char* {
    off = (off + 255) & ~(size_t)255;
    char* p = ws + off;
    off += bytes;
    return p;
  };
  float*  deg    = (float*)take((size_t)N_NODES * 4);
  float*  agg    = (float*)take((size_t)N_NODES * DD * 4);
  float*  xA     = (float*)take((size_t)N_NODES * DD * 4);
  float*  xB     = (float*)take((size_t)N_NODES * DD * 4);
  float*  xpre   = (float*)take((size_t)N_NODES * DD * 4);
  float*  ef_mid = (float*)take((size_t)N_EDGES * DD * 4);
  __bf16* fself  = (__bf16*)take((size_t)LAYERS * 4096 * 2);
  __bf16* fnbr   = (__bf16*)take((size_t)LAYERS * 4096 * 2);
  __bf16* ffc    = (__bf16*)take((size_t)LAYERS * 16384 * 2);
  float*  w256   = (float*)take((size_t)LAYERS * DD * 4);

  // degree (once)
  zero_f32<<<(N_NODES + 255) / 256, 256, 0, stream>>>(deg, N_NODES);
  compute_deg<<<(N_EDGES + 255) / 256, 256, 0, stream>>>(trg, deg, N_EDGES);
  // weight swizzle (once)
  const int prep_threads = LAYERS * 2 * 4 * 32 * 16 * 2 + LAYERS * 8 * 4 * 32 * 16 + LAYERS * DD;
  prep_weights<<<(prep_threads + 255) / 256, 256, 0, stream>>>(Wself, Wnbr, Wfc,
                                                               fself, fnbr, ffc, w256);

  const float* xcur[LAYERS]  = { x_in, xA, xB };
  float*       xrel[LAYERS]  = { xA, xB, xA };
  const float* efi[LAYERS]   = { ef0, out, ef_mid };
  float*       efo[LAYERS]   = { out, ef_mid, out };

  const int ntiles_n = N_NODES / 16;  // 3125
  for (int i = 0; i < LAYERS; ++i) {
    zero_f32<<<((N_NODES * DD) + 255) / 256, 256, 0, stream>>>(agg, N_NODES * DD);
    scatter_agg<<<N_EDGES / 16, 256, 0, stream>>>(xcur[i], src, trg, agg, N_EDGES);
    node_gemm<<<(ntiles_n + 3) / 4, 128, 0, stream>>>(
        xcur[i], agg, deg, fself + (size_t)i * 4096, fnbr + (size_t)i * 4096,
        bconv + i * DD, xpre, xrel[i], ntiles_n);
    edge_update<<<(N_EDGES / 16) / 4, 128, 0, stream>>>(
        xpre, src, trg, efi[i], ffc + (size_t)i * 16384, w256 + i * DD,
        bfc + i * DD, efo[i], N_EDGES);
  }
}
